// MultiHeadAttention_9878424781007
// MI455X (gfx1250) — compile-verified, matrix operand order checked by reading, not run on hardware
//
#include <hip/hip_runtime.h>
#include <hip/hip_bf16.h>
#include <math.h>
#include <stdint.h>

typedef __bf16 bf16;
typedef __attribute__((ext_vector_type(16))) __bf16 v16bf;
typedef __attribute__((ext_vector_type(8)))  __bf16 v8bf;
typedef __attribute__((ext_vector_type(4)))  __bf16 v4bf;
typedef __attribute__((ext_vector_type(8)))  float  v8f;
typedef __attribute__((ext_vector_type(4)))  float  v4f;

#define D_MODEL 1024
#define N_HEADS 16
#define D_K     64
#define BATCH   2
#define SEQ     2048
#define MTOT    (BATCH * SEQ)   /* 4096 */

// ---------------------------------------------------------------------------
// WMMA fragment load (wave32, 16x16x32 bf16): two ds_load_b128 per fragment.
//   lane<16 : idx = lane,    K = [0..7]  and [16..23]   (k0-relative)
//   lane>=16: idx = lane-16, K = [8..15] and [24..31]
// Valid for A from row-major [rows x K] and B from transposed storage Bt[n][k].
// ld must be a multiple of 8 elements (16B alignment).
// ---------------------------------------------------------------------------
__device__ __forceinline__ v16bf load_frag(const bf16* p, int idx0, int k0, int ld) {
  const int lane = threadIdx.x & 31;
  const int half = lane >> 4;
  const int m    = lane & 15;
  const bf16* rp = p + (idx0 + m) * ld + k0 + 8 * half;
  v8bf lo = *(const v8bf*)rp;
  v8bf hi = *(const v8bf*)(rp + 16);
  return __builtin_shufflevector(lo, hi, 0, 1, 2, 3, 4, 5, 6, 7,
                                 8, 9, 10, 11, 12, 13, 14, 15);
}

__device__ __forceinline__ v8f wmma_bf16(v16bf a, v16bf b, v8f c) {
  return __builtin_amdgcn_wmma_f32_16x16x32_bf16(false, a, false, b, (short)0, c,
                                                 false, false);
}

// ---------------------------------------------------------------------------
// Elementwise f32 -> bf16 conversion (prepass so GEMM A tiles are async-DMA'd)
// ---------------------------------------------------------------------------
__global__ __launch_bounds__(256)
void convert_bf16_kernel(const float* __restrict__ src, bf16* __restrict__ dst) {
  int i = blockIdx.x * 256 + threadIdx.x;          // one float4 per thread
  v4f x = *(const v4f*)(src + (size_t)i * 4);
  v4bf y;
  y[0] = (bf16)x[0]; y[1] = (bf16)x[1]; y[2] = (bf16)x[2]; y[3] = (bf16)x[3];
  *(v4bf*)(dst + (size_t)i * 4) = y;
}

// ---------------------------------------------------------------------------
// GEMM + bias: C[M,N] = A[M,K](bf16) @ W[K,N](f32) + bias, f32 accumulate.
// Block tile 256x128, 256 threads = 8 waves (4x2), wave tile 64x64.
// DOUBLE-BUFFERED pipeline: async-DMA A tile (i+1) + VGPR-stage W tile (i+1)
// while the 16 WMMAs of tile i run; single barrier per k-step.
// ---------------------------------------------------------------------------
#define GLDA 40  /* padded k-stride (elements) for LDS tiles */

template <typename TOut>
__global__ __launch_bounds__(256)
void gemm_bias_kernel(const bf16* __restrict__ A, const float* __restrict__ W,
                      const float* __restrict__ bias, TOut* __restrict__ C,
                      int M, int N, int K) {
  __shared__ alignas(16) bf16 As[2][256 * GLDA];   // row-major [row][k]
  __shared__ alignas(16) bf16 Bt[2][128 * GLDA];   // transposed [n][k]

  const int m0   = blockIdx.y * 256;
  const int n0   = blockIdx.x * 128;
  const int tid  = threadIdx.x;
  const int wave = tid >> 5;
  const int wrow0 = (wave >> 1) * 64;  // 0,64,128,192
  const int wcol0 = (wave & 1) * 64;   // 0,64
  const int lane = tid & 31;
  const int half = lane >> 4;
  const int nlo  = lane & 15;
  const int NK   = K >> 5;

  auto asyncA = [&](int kk, int buf) {
    const uint64_t gbase = (uint64_t)(uintptr_t)(A + (size_t)m0 * K + kk);
    const unsigned lbase = (unsigned)(uintptr_t)&As[buf][0];
#pragma unroll
    for (int j = 0; j < 4; ++j) {
      int idx = tid + j * 256;                 // 0..1023 chunk id
      int r = idx >> 2, ic = idx & 3;          // row, 16B chunk in row
      unsigned voff   = (unsigned)(r * (K * 2) + ic * 16);
      unsigned ldsoff = lbase + (unsigned)(r * (GLDA * 2) + ic * 16);
      asm volatile("global_load_async_to_lds_b128 %0, %1, %2"
                   :: "v"(ldsoff), "v"(voff), "s"(gbase) : "memory");
    }
  };
  auto loadW = [&](int kk, v4f w[4]) {
#pragma unroll
    for (int j = 0; j < 4; ++j) {
      int idx = tid + j * 256;                 // 0..1023 float4 id
      int row = idx >> 5, c4 = (idx & 31) * 4;
      w[j] = *(const v4f*)&W[(size_t)(kk + row) * N + n0 + c4];
    }
  };
  auto storeW = [&](int buf, const v4f w[4]) {
    bf16* bt = &Bt[buf][0];
#pragma unroll
    for (int j = 0; j < 4; ++j) {
      int idx = tid + j * 256;
      int row = idx >> 5, c4 = (idx & 31) * 4;
      bt[(c4 + 0) * GLDA + row] = (bf16)w[j][0];
      bt[(c4 + 1) * GLDA + row] = (bf16)w[j][1];
      bt[(c4 + 2) * GLDA + row] = (bf16)w[j][2];
      bt[(c4 + 3) * GLDA + row] = (bf16)w[j][3];
    }
  };

  v8f acc[4][4];
#pragma unroll
  for (int i = 0; i < 4; ++i)
#pragma unroll
    for (int j = 0; j < 4; ++j)
#pragma unroll
      for (int r = 0; r < 8; ++r) acc[i][j][r] = 0.0f;

  // prologue: stage tile 0
  asyncA(0, 0);
  {
    v4f w[4];
    loadW(0, w);
    storeW(0, w);
  }
  asm volatile("s_wait_asynccnt 0" ::: "memory");
  __syncthreads();

  for (int i = 0; i < NK; ++i) {
    const int cur  = i & 1, nxt = cur ^ 1;
    const bool more = (i + 1 < NK);
    v4f w[4];
    if (more) {
      asyncA((i + 1) * 32, nxt);     // DMA next A tile during compute
      loadW((i + 1) * 32, w);        // next W tile into VGPRs
    }

    v16bf af[4], bfr[4];
#pragma unroll
    for (int t = 0; t < 4; ++t) af[t]  = load_frag(&As[cur][0], wrow0 + 16 * t, 0, GLDA);
#pragma unroll
    for (int t = 0; t < 4; ++t) bfr[t] = load_frag(&Bt[cur][0], wcol0 + 16 * t, 0, GLDA);
#pragma unroll
    for (int ti = 0; ti < 4; ++ti)
#pragma unroll
      for (int tj = 0; tj < 4; ++tj)
        acc[ti][tj] = wmma_bf16(af[ti], bfr[tj], acc[ti][tj]);

    if (more) {
      storeW(nxt, w);
      asm volatile("s_wait_asynccnt 0" ::: "memory");
    }
    __syncthreads();
  }

  // epilogue: bias + store
  float bj[4];
#pragma unroll
  for (int j = 0; j < 4; ++j) bj[j] = bias[n0 + wcol0 + 16 * j + nlo];
#pragma unroll
  for (int i = 0; i < 4; ++i)
#pragma unroll
    for (int j = 0; j < 4; ++j)
#pragma unroll
      for (int r = 0; r < 8; ++r) {
        int row = m0 + wrow0 + 16 * i + r + 8 * half;
        int col = n0 + wcol0 + 16 * j + nlo;
        C[(size_t)row * N + col] = (TOut)(acc[i][j][r] + bj[j]);
      }
}

// ---------------------------------------------------------------------------
// Flash attention (causal). Grid: (B*H, S/64). 128 threads = 4 waves.
// K tile double-buffered via async DMA with partial-count waits
// (s_wait_asynccnt 4 keeps the next tile's 4 chunks in flight).
// ---------------------------------------------------------------------------
#define QLD 72   /* padded dk-stride for Qs/Ks/Ps/Vt */
#define SLD 68   /* padded key-stride for Ss (floats) */

__global__ __launch_bounds__(128)
void attention_kernel(const bf16* __restrict__ Qp, const bf16* __restrict__ Kp,
                      const bf16* __restrict__ Vp, bf16* __restrict__ AO) {
  __shared__ alignas(16) bf16  Qs[64 * QLD];      // [q][dk]
  __shared__ alignas(16) bf16  Ks[2][64 * QLD];   // [key][dk], double buffered
  __shared__ alignas(16) bf16  Vt[64 * QLD];      // [dk][key]
  __shared__ alignas(16) float Ss[64 * SLD];      // scores [q][key]
  __shared__ alignas(16) bf16  Ps[64 * QLD];      // probs  [q][key]
  __shared__ float rowM[64], rowL[64], rowScale[64];

  const int bh   = blockIdx.x;             // 0..31
  const int b    = bh / N_HEADS;
  const int h    = bh % N_HEADS;
  const int qb   = blockIdx.y;             // 0..31
  const int tid  = threadIdx.x;
  const int wave = tid >> 5;               // 0..3
  const int lane = tid & 31;
  const int half = lane >> 4;
  const int nlo  = lane & 15;

  const size_t qbase = (size_t)(b * SEQ + qb * 64) * D_MODEL + h * D_K;

  auto asyncK = [&](int kb, int buf) {
    const size_t kvbase = (size_t)(b * SEQ + kb * 64) * D_MODEL + h * D_K;
    const uint64_t gbase = (uint64_t)(uintptr_t)(Kp + kvbase);
    const unsigned lbase = (unsigned)(uintptr_t)&Ks[buf][0];
#pragma unroll
    for (int j = 0; j < 4; ++j) {
      int c = tid + j * 128;
      int r = c >> 3, ib = (c & 7) * 16;   // row, byte chunk in row
      unsigned voff   = (unsigned)(r * (D_MODEL * 2) + ib);
      unsigned ldsoff = lbase + (unsigned)(r * (QLD * 2) + ib);
      asm volatile("global_load_async_to_lds_b128 %0, %1, %2"
                   :: "v"(ldsoff), "v"(voff), "s"(gbase) : "memory");
    }
  };

  // Q tile: 16B copies (uniform loop), plus prefetch K(0)
  asyncK(0, 0);
#pragma unroll
  for (int j = 0; j < 4; ++j) {
    int c = tid + j * 128;
    int r = c >> 3, i8 = (c & 7) * 8;
    *(v8bf*)&Qs[r * QLD + i8] = *(const v8bf*)&Qp[qbase + (size_t)r * D_MODEL + i8];
  }
  if (tid < 64) { rowM[tid] = -1e30f; rowL[tid] = 0.0f; }
  __syncthreads();

  v16bf aQ[2];
#pragma unroll
  for (int ks = 0; ks < 2; ++ks) aQ[ks] = load_frag(Qs, wave * 16, ks * 32, QLD);

  v8f oacc[4];
#pragma unroll
  for (int j = 0; j < 4; ++j)
#pragma unroll
    for (int r = 0; r < 8; ++r) oacc[j][r] = 0.0f;

  const float sm_scale = 0.125f;  // 1/sqrt(64)

  for (int kb = 0; kb <= qb; ++kb) {
    const int cur = kb & 1, nxt = cur ^ 1;
    const size_t kvbase = (size_t)(b * SEQ + kb * 64) * D_MODEL + h * D_K;

    if (kb < qb) asyncK(kb + 1, nxt);      // prefetch next K tile

    // V tile: 16B row reads, transposed scalar stores
#pragma unroll
    for (int j = 0; j < 4; ++j) {
      int c = tid + j * 128;
      int r = c >> 3, i8 = (c & 7) * 8;
      v8bf t = *(const v8bf*)&Vp[kvbase + (size_t)r * D_MODEL + i8];
#pragma unroll
      for (int i = 0; i < 8; ++i) Vt[(i8 + i) * QLD + r] = t[i];
    }

    // ASYNCcnt completes in order: <=4 outstanding proves K(kb) has landed
    // while K(kb+1)'s 4 chunks stay in flight.
    if (kb < qb) {
      asm volatile("s_wait_asynccnt 4" ::: "memory");
    } else {
      asm volatile("s_wait_asynccnt 0" ::: "memory");
    }
    __syncthreads();

    // S = (Q K^T) * scale with causal mask -> Ss
    const bf16* kcur = &Ks[cur][0];
#pragma unroll
    for (int nt = 0; nt < 4; ++nt) {
      v8f s;
#pragma unroll
      for (int r = 0; r < 8; ++r) s[r] = 0.0f;
#pragma unroll
      for (int ks = 0; ks < 2; ++ks)
        s = wmma_bf16(aQ[ks], load_frag(kcur, nt * 16, ks * 32, QLD), s);
#pragma unroll
      for (int r = 0; r < 8; ++r) {
        int lrow = wave * 16 + r + 8 * half;
        int lcol = nt * 16 + nlo;
        int gq = qb * 64 + lrow;
        int gk = kb * 64 + lcol;
        Ss[lrow * SLD + lcol] = (gk <= gq) ? s[r] * sm_scale : -1e9f;
      }
    }
    __syncthreads();

    // online softmax: lanes 0..15 of each wave own one row; float4 LDS I/O
    if (lane < 16) {
      int row = wave * 16 + lane;
      const v4f* srow = (const v4f*)&Ss[row * SLD];
      float mold = rowM[row];
      float mnew = mold;
#pragma unroll
      for (int c4 = 0; c4 < 16; ++c4) {
        v4f x = srow[c4];
        mnew = fmaxf(mnew, fmaxf(fmaxf(x[0], x[1]), fmaxf(x[2], x[3])));
      }
      float alpha = __expf(mold - mnew);
      float sum = 0.0f;
#pragma unroll
      for (int c4 = 0; c4 < 16; ++c4) {
        v4f x = srow[c4];
        float p0 = __expf(x[0] - mnew), p1 = __expf(x[1] - mnew);
        float p2 = __expf(x[2] - mnew), p3 = __expf(x[3] - mnew);
        v4bf pv;
        pv[0] = (bf16)p0; pv[1] = (bf16)p1; pv[2] = (bf16)p2; pv[3] = (bf16)p3;
        *(v4bf*)&Ps[row * QLD + c4 * 4] = pv;
        sum += (p0 + p1) + (p2 + p3);
      }
      rowL[row]     = rowL[row] * alpha + sum;
      rowM[row]     = mnew;
      rowScale[row] = alpha;
    }
    __syncthreads();

    // rescale accumulators, then O += P @ V
    float sc[8];
#pragma unroll
    for (int r = 0; r < 8; ++r) sc[r] = rowScale[wave * 16 + r + 8 * half];
#pragma unroll
    for (int j = 0; j < 4; ++j)
#pragma unroll
      for (int r = 0; r < 8; ++r) oacc[j][r] *= sc[r];

    v16bf aP[2];
#pragma unroll
    for (int ks = 0; ks < 2; ++ks) aP[ks] = load_frag(Ps, wave * 16, ks * 32, QLD);
#pragma unroll
    for (int nt = 0; nt < 4; ++nt)
#pragma unroll
      for (int ks = 0; ks < 2; ++ks)
        oacc[nt] = wmma_bf16(aP[ks], load_frag(Vt, nt * 16, ks * 32, QLD), oacc[nt]);
    __syncthreads();
  }

  // normalize and store attention output (bf16, [B*S, D], head columns)
  float invl[8];
#pragma unroll
  for (int r = 0; r < 8; ++r) invl[r] = 1.0f / rowL[wave * 16 + r + 8 * half];
#pragma unroll
  for (int nt = 0; nt < 4; ++nt)
#pragma unroll
    for (int r = 0; r < 8; ++r) {
      int lrow = wave * 16 + r + 8 * half;
      int lcol = nt * 16 + nlo;
      AO[qbase + (size_t)lrow * D_MODEL + lcol] = (bf16)(oacc[nt][r] * invl[r]);
    }
}

// ---------------------------------------------------------------------------
// launch: Ab buffer is reused for q/k/v bf16 conversions and then aliased as
// the attention output (sequential on one stream), keeping ws at 32 MB:
//   Ab/AO | Qp | Kp | Vp   (each bf16 4096x1024 = 8 MB)
// ---------------------------------------------------------------------------
extern "C" void kernel_launch(void* const* d_in, const int* in_sizes, int n_in,
                              void* d_out, int out_size, void* d_ws, size_t ws_size,
                              hipStream_t stream) {
  const float* q   = (const float*)d_in[0];
  const float* k   = (const float*)d_in[1];
  const float* v   = (const float*)d_in[2];
  /* d_in[3] = causal mask, implicit */
  const float* W_q = (const float*)d_in[4];
  const float* b_q = (const float*)d_in[5];
  const float* W_k = (const float*)d_in[6];
  const float* b_k = (const float*)d_in[7];
  const float* W_v = (const float*)d_in[8];
  const float* b_v = (const float*)d_in[9];
  const float* W_o = (const float*)d_in[10];
  const float* b_o = (const float*)d_in[11];
  float* out = (float*)d_out;

  bf16* Ab = (bf16*)d_ws;                      // conversion buffer / AO alias
  bf16* Qp = Ab + (size_t)MTOT * D_MODEL;
  bf16* Kp = Qp + (size_t)MTOT * D_MODEL;
  bf16* Vp = Kp + (size_t)MTOT * D_MODEL;

  const int cgrid = (MTOT * D_MODEL / 4) / 256;     // 4096 blocks
  dim3 ggrid(D_MODEL / 128, MTOT / 256);            // (8, 16)

  convert_bf16_kernel<<<cgrid, 256, 0, stream>>>(q, Ab);
  gemm_bias_kernel<bf16><<<ggrid, 256, 0, stream>>>(Ab, W_q, b_q, Qp, MTOT, D_MODEL, D_MODEL);
  convert_bf16_kernel<<<cgrid, 256, 0, stream>>>(k, Ab);
  gemm_bias_kernel<bf16><<<ggrid, 256, 0, stream>>>(Ab, W_k, b_k, Kp, MTOT, D_MODEL, D_MODEL);
  convert_bf16_kernel<<<cgrid, 256, 0, stream>>>(v, Ab);
  gemm_bias_kernel<bf16><<<ggrid, 256, 0, stream>>>(Ab, W_v, b_v, Vp, MTOT, D_MODEL, D_MODEL);

  attention_kernel<<<dim3(BATCH * N_HEADS, SEQ / 64), 128, 0, stream>>>(Qp, Kp, Vp, Ab);

  gemm_bias_kernel<float><<<ggrid, 256, 0, stream>>>(Ab, W_o, b_o, out, MTOT, D_MODEL, D_MODEL);
}
